// GCN_67336497266833
// MI455X (gfx1250) — compile-verified, matrix operand order checked by reading, not run on hardware
//
#include <hip/hip_runtime.h>
#include <hip/hip_bf16.h>

typedef __attribute__((ext_vector_type(16))) __bf16 v16bf;
typedef __attribute__((ext_vector_type(8)))  float  v8f;

#define N_NODES 50000
#define N_EDGES 800000
#define DIM_IN  256
#define DIM_H   256
#define DIM_OUT 128

// ---------------------------------------------------------------------------
// Degree / normalization kernels
// ---------------------------------------------------------------------------
__global__ void deg_zero_kernel(float* __restrict__ deg, int n) {
    int i = blockIdx.x * blockDim.x + threadIdx.x;
    if (i < n) deg[i] = 0.0f;
}

__global__ void deg_count_kernel(const int* __restrict__ dst,
                                 float* __restrict__ deg, int ne) {
    int e = blockIdx.x * blockDim.x + threadIdx.x;
    if (e < ne) {
        int d = dst[e];
        if (d >= 0 && d < N_NODES) atomicAdd(&deg[d], 1.0f);
    }
}

__global__ void dinv_kernel(float* __restrict__ deg, int n) {
    int i = blockIdx.x * blockDim.x + threadIdx.x;
    if (i < n) deg[i] = rsqrtf(deg[i] + 1.0f);   // deg of A + I
}

// ---------------------------------------------------------------------------
// Fused (gather | relu) + GEMM with bf16 WMMA, f32 accumulate.
//   H[m, n] = sum_k act(Arow(m)[k]) * W[k, n]
// Block: 256 threads = 8 wave32s; block tile = 64 rows x 32 cols;
// each wave owns one 16x16 WMMA tile (4 row-tiles x 2 col-tiles).
// K, N are compile-time so all strides become immediate offsets.
// ---------------------------------------------------------------------------
template <bool GATHER, bool RELU, int K, int N>
__global__ void gemm_wmma_kernel(const float* __restrict__ A,
                                 const int* __restrict__ idx,
                                 const float* __restrict__ W,
                                 float* __restrict__ H,
                                 int M) {
    __shared__ __bf16 As[64][34];   // [row][k]   (+pad vs bank conflicts)
    __shared__ __bf16 Bs[32][34];   // [n][k]     (transposed store of W tile)

    const int tid  = threadIdx.x;
    const int lane = tid & 31;
    const int wave = tid >> 5;

    const int row_base = blockIdx.x * 64;
    const int col_base = blockIdx.y * 32;
    const int row_sub  = (wave >> 1) << 4;   // 0,16,32,48
    const int col_sub  = (wave & 1) << 4;    // 0,16

    // A-tile loader mapping: 64 rows x 32 cols, 8 floats per thread
    const int a_r = tid >> 2;          // 0..63
    const int a_c = (tid & 3) << 3;    // 0,8,16,24
    int grow = row_base + a_r;
    int srow;
    if (GATHER) {
        srow = (grow < M) ? idx[grow] : 0;
        if (srow < 0 || srow >= M) srow = 0;   // defensive clamp
    } else {
        srow = (grow < M) ? grow : 0;
    }
    const float* __restrict__ arow = A + (size_t)srow * K;

    // B-tile loader mapping: 32 k-rows x 32 n-cols, 4 floats per thread
    const int b_k = tid >> 3;          // 0..31
    const int b_n = (tid & 7) << 2;    // 0,4,...,28
    const float* __restrict__ wrow0 = W + (size_t)b_k * N + col_base + b_n;

    v8f acc = {0.f, 0.f, 0.f, 0.f, 0.f, 0.f, 0.f, 0.f};

    const int mloc = row_sub + (lane & 15);
    const int nloc = col_sub + (lane & 15);
    const int ksel = (lane & 16) ? 8 : 0;   // ISA 16-bit A/B lane->K mapping

    for (int k0 = 0; k0 < K; k0 += 32) {
        // stage A tile (f32 -> bf16)
        float4 av0 = *(const float4*)(arow + k0 + a_c);
        float4 av1 = *(const float4*)(arow + k0 + a_c + 4);
        if (RELU) {
            av0.x = fmaxf(av0.x, 0.f); av0.y = fmaxf(av0.y, 0.f);
            av0.z = fmaxf(av0.z, 0.f); av0.w = fmaxf(av0.w, 0.f);
            av1.x = fmaxf(av1.x, 0.f); av1.y = fmaxf(av1.y, 0.f);
            av1.z = fmaxf(av1.z, 0.f); av1.w = fmaxf(av1.w, 0.f);
        }
        As[a_r][a_c + 0] = (__bf16)av0.x;
        As[a_r][a_c + 1] = (__bf16)av0.y;
        As[a_r][a_c + 2] = (__bf16)av0.z;
        As[a_r][a_c + 3] = (__bf16)av0.w;
        As[a_r][a_c + 4] = (__bf16)av1.x;
        As[a_r][a_c + 5] = (__bf16)av1.y;
        As[a_r][a_c + 6] = (__bf16)av1.z;
        As[a_r][a_c + 7] = (__bf16)av1.w;

        // stage B tile transposed: Bs[n][k] = W[k0+k][col_base+n]
        float4 bv = *(const float4*)(wrow0 + (size_t)k0 * N);
        Bs[b_n + 0][b_k] = (__bf16)bv.x;
        Bs[b_n + 1][b_k] = (__bf16)bv.y;
        Bs[b_n + 2][b_k] = (__bf16)bv.z;
        Bs[b_n + 3][b_k] = (__bf16)bv.w;

        __syncthreads();

        // Build fragments per CDNA5 16-bit 16x32 layout:
        // lanes 0-15: K 0..7 / 16..23 ; lanes 16-31: K 8..15 / 24..31
        v16bf afrag, bfrag;
#pragma unroll
        for (int e = 0; e < 8; ++e) {
            afrag[e]     = As[mloc][ksel + e];
            afrag[e + 8] = As[mloc][16 + ksel + e];
            bfrag[e]     = Bs[nloc][ksel + e];
            bfrag[e + 8] = Bs[nloc][16 + ksel + e];
        }

        acc = __builtin_amdgcn_wmma_f32_16x16x32_bf16(
            /*neg_a=*/false, afrag, /*neg_b=*/false, bfrag,
            /*c_mod=*/(short)0, acc, /*reuse_a=*/false, /*reuse_b=*/false);

        __syncthreads();
    }

    // Writeback per f32 C/D layout: lane<16 -> M=v, lane>=16 -> M=v+8.
    // Uniform fast path when whole 64-row block tile is in bounds.
    const int n_out   = col_base + col_sub + (lane & 15);
    const int m0      = (lane & 16) ? 8 : 0;
    const int m_start = row_base + row_sub + m0;
    float* __restrict__ o = H + (size_t)m_start * N + n_out;
    if (row_base + 64 <= M) {
#pragma unroll
        for (int v = 0; v < 8; ++v) o[(size_t)v * N] = acc[v];
    } else {
#pragma unroll
        for (int v = 0; v < 8; ++v)
            if (m_start + v < M) o[(size_t)v * N] = acc[v];
    }
}

// ---------------------------------------------------------------------------
// out[i, c] = h[i, c] * dinv[i]^2 + b[c]   (full init: no pre-zero needed)
// ---------------------------------------------------------------------------
template <int NCOLS>
__global__ void self_init_kernel(const float* __restrict__ h,
                                 const float* __restrict__ dinv,
                                 const float* __restrict__ bias,
                                 float* __restrict__ out,
                                 int n) {
    size_t g = (size_t)blockIdx.x * blockDim.x + threadIdx.x;
    size_t total = (size_t)n * NCOLS;
    if (g < total) {
        int i = (int)(g / NCOLS);          // NCOLS is pow2 -> shift
        int c = (int)(g & (NCOLS - 1));
        float di = dinv[i];
        out[g] = h[g] * di * di + bias[c];
    }
}

// ---------------------------------------------------------------------------
// Edge scatter-add:  out[dst] += h[src] * dinv[src]*dinv[dst]
// NCOLS/4 threads per edge, float4 gather + 4x f32 atomics.
// ---------------------------------------------------------------------------
template <int NCOLS>
__global__ void edge_agg_kernel(const float* __restrict__ h,
                                const int* __restrict__ src,
                                const int* __restrict__ dst,
                                const float* __restrict__ dinv,
                                float* __restrict__ out,
                                int ne) {
    constexpr int TPE = NCOLS >> 2;  // threads per edge (pow2)
    long long g = (long long)blockIdx.x * blockDim.x + threadIdx.x;
    int e  = (int)(g / TPE);         // constexpr divisor -> shift
    int c4 = ((int)g & (TPE - 1)) << 2;
    if (e < ne) {
        int s = src[e];
        int d = dst[e];
        if (s < 0 || s >= N_NODES || d < 0 || d >= N_NODES) return;
        float norm = dinv[s] * dinv[d];
        float4 v = *(const float4*)(h + (size_t)s * NCOLS + c4);
        float* o = out + (size_t)d * NCOLS + c4;
        atomicAdd(o + 0, v.x * norm);
        atomicAdd(o + 1, v.y * norm);
        atomicAdd(o + 2, v.z * norm);
        atomicAdd(o + 3, v.w * norm);
    }
}

// ---------------------------------------------------------------------------
extern "C" void kernel_launch(void* const* d_in, const int* in_sizes, int n_in,
                              void* d_out, int out_size, void* d_ws, size_t ws_size,
                              hipStream_t stream) {
    (void)in_sizes; (void)n_in; (void)out_size; (void)ws_size;

    const int*   feat_idx  = (const int*)d_in[0];
    const int*   edge_list = (const int*)d_in[1];
    const float* emb       = (const float*)d_in[2];
    const float* W1        = (const float*)d_in[3];
    const float* b1        = (const float*)d_in[4];
    const float* W2        = (const float*)d_in[5];
    const float* b2        = (const float*)d_in[6];
    float*       out       = (float*)d_out;
    float*       ws        = (float*)d_ws;

    const int* src = edge_list;             // edge_list[0, :]
    const int* dst = edge_list + N_EDGES;   // edge_list[1, :]

    // Workspace layout (floats, 16B-aligned offsets):
    float* dinv = ws;                                        // 50000
    float* h1   = ws + 50176;                                // 50000*256
    float* agg1 = ws + 50176 + (size_t)N_NODES * DIM_H;      // 50000*256
    float* h2   = h1;  // h1 region is free once layer-1 aggregation is done

    const int T = 256;

    // 1) degrees -> dinv (in place)
    deg_zero_kernel<<<(N_NODES + T - 1) / T, T, 0, stream>>>(dinv, N_NODES);
    deg_count_kernel<<<(N_EDGES + T - 1) / T, T, 0, stream>>>(dst, dinv, N_EDGES);
    dinv_kernel<<<(N_NODES + T - 1) / T, T, 0, stream>>>(dinv, N_NODES);

    // 2) h1 = emb[feat_idx] @ W1   (gather-fused WMMA GEMM)
    dim3 g1((N_NODES + 63) / 64, DIM_H / 32);
    gemm_wmma_kernel<true, false, DIM_IN, DIM_H><<<g1, T, 0, stream>>>(
        emb, feat_idx, W1, h1, N_NODES);

    // 3) agg1 = h1*dinv^2 + b1 ; then scatter-add edge messages
    size_t tot1 = (size_t)N_NODES * DIM_H;
    self_init_kernel<DIM_H><<<(unsigned)((tot1 + T - 1) / T), T, 0, stream>>>(
        h1, dinv, b1, agg1, N_NODES);
    long long ethr1 = (long long)N_EDGES * (DIM_H / 4);
    edge_agg_kernel<DIM_H><<<(unsigned)((ethr1 + T - 1) / T), T, 0, stream>>>(
        h1, src, dst, dinv, agg1, N_EDGES);

    // 4) h2 = relu(agg1) @ W2   (ReLU fused into A-tile load)
    dim3 g2((N_NODES + 63) / 64, DIM_OUT / 32);
    gemm_wmma_kernel<false, true, DIM_H, DIM_OUT><<<g2, T, 0, stream>>>(
        agg1, nullptr, W2, h2, N_NODES);

    // 5) out = h2*dinv^2 + b2 ; then scatter-add edge messages into d_out
    size_t tot2 = (size_t)N_NODES * DIM_OUT;
    self_init_kernel<DIM_OUT><<<(unsigned)((tot2 + T - 1) / T), T, 0, stream>>>(
        h2, dinv, b2, out, N_NODES);
    long long ethr2 = (long long)N_EDGES * (DIM_OUT / 4);
    edge_agg_kernel<DIM_OUT><<<(unsigned)((ethr2 + T - 1) / T), T, 0, stream>>>(
        h2, src, dst, dinv, out, N_EDGES);
}
